// MaskedMultiHeadAttention_66365834658467
// MI455X (gfx1250) — compile-verified, matrix operand order checked by reading, not run on hardware
//
#include <hip/hip_runtime.h>

#define BB 2
#define LL 4096
#define DD 512
#define HH 8
#define EE 64

typedef __attribute__((ext_vector_type(16))) _Float16 v16h;
typedef __attribute__((ext_vector_type(8)))  _Float16 v8h;
typedef __attribute__((ext_vector_type(8)))  float    v8f;

union frag16 { v16h v; v8h h[2]; };

__device__ __forceinline__ v8f wmma_f16(v16h a, v16h b, v8f c) {
  // 8 args: (neg_a, A, neg_b, B, c_mod, C, reuse_a, reuse_b)
  return __builtin_amdgcn_wmma_f32_16x16x32_f16(false, a, false, b, (short)0, c, false, false);
}

// Async global->LDS copy, 16B per lane, tracked by ASYNCcnt (CDNA5 TDM-lite path).
__device__ __forceinline__ void async_b128(const _Float16* lds_dst, const _Float16* gsrc) {
  unsigned lds_off = (unsigned)(size_t)lds_dst;  // low 32 bits of LDS-aperture addr = LDS byte addr
  asm volatile("global_load_async_to_lds_b128 %0, %1, off"
               :: "v"(lds_off), "v"(gsrc) : "memory");
}
__device__ __forceinline__ void wait_async0() {
  asm volatile("s_wait_asynccnt 0x0" ::: "memory");
}

// Load a 16x32 f16 A-fragment (or 32x16 B-fragment) from a row-major tile.
// ISA layout: lane&15 = row (A) / col (B); lane>>4 picks K-half:
// v16h elems 0..7 = K[kb..kb+7], elems 8..15 = K[kb+16..kb+23], kb = (lane>>4)*8.
__device__ __forceinline__ v16h load_frag(const _Float16* base, int stride, int lane) {
  const _Float16* p = base + (lane & 15) * stride + ((lane >> 4) << 3);
  frag16 f;
  f.h[0] = *(const v8h*)(p);
  f.h[1] = *(const v8h*)(p + 16);
  return f.v;
}

__device__ __forceinline__ float redmax16(float v) {
#pragma unroll
  for (int off = 1; off < 16; off <<= 1) v = fmaxf(v, __shfl_xor(v, off, 32));
  return v;
}
__device__ __forceinline__ float redsum16(float v) {
#pragma unroll
  for (int off = 1; off < 16; off <<= 1) v += __shfl_xor(v, off, 32);
  return v;
}

// ---------------- weight transpose + f16 convert: Wt[n][k] = (f16)W[k][n] ----------------
__global__ __launch_bounds__(256) void k_wT(const float* __restrict__ W, _Float16* __restrict__ Wt) {
  int idx = blockIdx.x * 256 + threadIdx.x;
  int n = idx >> 9, k = idx & 511;
  Wt[n * DD + k] = (_Float16)W[k * DD + n];
}

// ---------------- projection GEMM: Y = X(fp32, MxK row-major) * Wt^T, f16 out ----------------
// Block tile 128x128, 256 threads = 8 waves (2x4), each wave 64x32 (4x2 WMMA tiles), K-step 32.
// A tile: fp32->f16 via VGPR (needs convert). B tile: async global->LDS. Double-buffered.
// vmode 0: Y[b][h][l][e]   vmode 1: Y[b][h][e][l] (transposed, for V)
__global__ __launch_bounds__(256) void k_proj(const float* __restrict__ X,
                                              const _Float16* __restrict__ Wt,
                                              _Float16* __restrict__ Y,
                                              float scale, int vmode) {
  __shared__ _Float16 As[2][128 * 40];
  __shared__ _Float16 Bs[2][128 * 40];
  const int t = threadIdx.x;
  const int lane = t & 31;
  const int w = t >> 5;
  const int wm = w >> 2, wn = w & 3;
  const int m0 = blockIdx.y * 128, n0 = blockIdx.x * 128;

  v8f acc[4][2];
  v8f zero = {};
#pragma unroll
  for (int i = 0; i < 4; ++i)
#pragma unroll
    for (int j = 0; j < 2; ++j) acc[i][j] = zero;

  auto stageA = [&](int buf, int k0) {
#pragma unroll
    for (int c = 0; c < 4; ++c) {
      int off = t * 16 + c * 4;
      int row = off >> 5, col = off & 31;
      float4 f = *(const float4*)(X + (size_t)(m0 + row) * DD + k0 + col);
      _Float16* d = &As[buf][row * 40 + col];
      d[0] = (_Float16)f.x; d[1] = (_Float16)f.y; d[2] = (_Float16)f.z; d[3] = (_Float16)f.w;
    }
  };
  auto stageB = [&](int buf, int k0) {
#pragma unroll
    for (int c = 0; c < 2; ++c) {
      int off = t * 16 + c * 8;
      int row = off >> 5, col = off & 31;
      async_b128(&Bs[buf][row * 40 + col], Wt + (size_t)(n0 + row) * DD + k0 + col);
    }
  };

  stageA(0, 0);
  stageB(0, 0);
  for (int k0 = 0; k0 < DD; k0 += 32) {
    const int buf = (k0 >> 5) & 1;
    wait_async0();
    __syncthreads();
    if (k0 + 32 < DD) {
      __builtin_prefetch(X + (size_t)(m0 + (t >> 1)) * DD + k0 + 32, 0, 1);
      stageA(buf ^ 1, k0 + 32);
      stageB(buf ^ 1, k0 + 32);
    }
    v16h a[4], bfr[2];
#pragma unroll
    for (int i = 0; i < 4; ++i) a[i] = load_frag(&As[buf][(wm * 64 + i * 16) * 40], 40, lane);
#pragma unroll
    for (int j = 0; j < 2; ++j) bfr[j] = load_frag(&Bs[buf][(wn * 32 + j * 16) * 40], 40, lane);
#pragma unroll
    for (int i = 0; i < 4; ++i)
#pragma unroll
      for (int j = 0; j < 2; ++j) acc[i][j] = wmma_f16(a[i], bfr[j], acc[i][j]);
  }

  const int rbase = (lane >> 4) << 3;
  const int cbase = lane & 15;
#pragma unroll
  for (int i = 0; i < 4; ++i)
#pragma unroll
    for (int j = 0; j < 2; ++j)
#pragma unroll
      for (int r = 0; r < 8; ++r) {
        int row = m0 + wm * 64 + i * 16 + r + rbase;
        int col = n0 + wn * 32 + j * 16 + cbase;
        float v = acc[i][j][r] * scale;
        int b = row >> 12, l = row & (LL - 1);
        int h = col >> 6, e = col & (EE - 1);
        if (vmode == 0)
          Y[(size_t)((b * HH + h) * LL + l) * EE + e] = (_Float16)v;
        else
          Y[(size_t)((b * HH + h) * EE + e) * LL + l] = (_Float16)v;
      }
}

// ---------------- output GEMM: out(fp32) = A(f16, Mx512 row-major) * Wot^T ----------------
// Fully async-staged, double-buffered.
__global__ __launch_bounds__(256) void k_oproj(const _Float16* __restrict__ A,
                                               const _Float16* __restrict__ Wt,
                                               float* __restrict__ out) {
  __shared__ _Float16 As[2][128 * 40];
  __shared__ _Float16 Bs[2][128 * 40];
  const int t = threadIdx.x;
  const int lane = t & 31;
  const int w = t >> 5;
  const int wm = w >> 2, wn = w & 3;
  const int m0 = blockIdx.y * 128, n0 = blockIdx.x * 128;

  v8f acc[4][2];
  v8f zero = {};
#pragma unroll
  for (int i = 0; i < 4; ++i)
#pragma unroll
    for (int j = 0; j < 2; ++j) acc[i][j] = zero;

  auto stage = [&](int buf, int k0) {
#pragma unroll
    for (int c = 0; c < 2; ++c) {
      int off = t * 16 + c * 8;
      int row = off >> 5, col = off & 31;
      async_b128(&As[buf][row * 40 + col], A  + (size_t)(m0 + row) * DD + k0 + col);
      async_b128(&Bs[buf][row * 40 + col], Wt + (size_t)(n0 + row) * DD + k0 + col);
    }
  };

  stage(0, 0);
  for (int k0 = 0; k0 < DD; k0 += 32) {
    const int buf = (k0 >> 5) & 1;
    wait_async0();
    __syncthreads();
    if (k0 + 32 < DD) stage(buf ^ 1, k0 + 32);
    v16h a[4], bfr[2];
#pragma unroll
    for (int i = 0; i < 4; ++i) a[i] = load_frag(&As[buf][(wm * 64 + i * 16) * 40], 40, lane);
#pragma unroll
    for (int j = 0; j < 2; ++j) bfr[j] = load_frag(&Bs[buf][(wn * 32 + j * 16) * 40], 40, lane);
#pragma unroll
    for (int i = 0; i < 4; ++i)
#pragma unroll
      for (int j = 0; j < 2; ++j) acc[i][j] = wmma_f16(a[i], bfr[j], acc[i][j]);
  }

  const int rbase = (lane >> 4) << 3;
  const int cbase = lane & 15;
#pragma unroll
  for (int i = 0; i < 4; ++i)
#pragma unroll
    for (int j = 0; j < 2; ++j)
#pragma unroll
      for (int r = 0; r < 8; ++r) {
        int row = m0 + wm * 64 + i * 16 + r + rbase;
        int col = n0 + wn * 32 + j * 16 + cbase;
        out[(size_t)row * DD + col] = acc[i][j][r];
      }
}

// ---------------- causal flash attention ----------------
// grid: (L/64, B*H); block: 128 threads = 4 waves; wave w owns query rows qb+w*16 .. +15.
// K/V tiles async-staged into double-buffered LDS.
__global__ __launch_bounds__(128) void k_attn(const _Float16* __restrict__ Qh,
                                              const _Float16* __restrict__ Kh,
                                              const _Float16* __restrict__ Vt,
                                              _Float16* __restrict__ Oc) {
  __shared__ _Float16 Ks[2][64 * 72];
  __shared__ _Float16 Vs[2][64 * 72];
  __shared__ _Float16 Ps[4 * 16 * 72];
  const int t = threadIdx.x;
  const int lane = t & 31;
  const int w = t >> 5;
  const int qb = blockIdx.x * 64;
  const int bh = blockIdx.y;
  const int b = bh >> 3, h = bh & 7;
  const int rbase = (lane >> 4) << 3;
  const int cbase = lane & 15;

  const _Float16* Qbase = Qh + ((size_t)bh * LL + qb + w * 16) * EE;
  v16h q0 = load_frag(Qbase, EE, lane);
  v16h q1 = load_frag(Qbase + 32, EE, lane);

  v8f zero = {};
  v8f o[4];
  float mr[8], sr[8];
  int rowg[8];
#pragma unroll
  for (int j = 0; j < 4; ++j) o[j] = zero;
#pragma unroll
  for (int r = 0; r < 8; ++r) { mr[r] = -1e30f; sr[r] = 0.f; rowg[r] = qb + w * 16 + r + rbase; }

  auto stageKV = [&](int buf, int s0) {
#pragma unroll
    for (int c = 0; c < 4; ++c) {
      int off = t * 32 + c * 8;
      int row = off >> 6, col = off & 63;
      async_b128(&Ks[buf][row * 72 + col], Kh + ((size_t)bh * LL + s0 + row) * EE + col);
      async_b128(&Vs[buf][row * 72 + col], Vt + ((size_t)bh * EE + row) * LL + s0 + col);
    }
  };

  const int nblk = (qb >> 6) + 1;
  stageKV(0, 0);
  for (int it = 0; it < nblk; ++it) {
    const int s0 = it << 6;
    const int buf = it & 1;
    wait_async0();
    __syncthreads();
    if (it + 1 < nblk) stageKV(buf ^ 1, s0 + 64);

    // S = Q * K^T  (B-fragment lane = key index, rows of Ks are keys -> contiguous)
    v8f sc[4];
#pragma unroll
    for (int j = 0; j < 4; ++j) {
      sc[j] = zero;
      v16h kb0 = load_frag(&Ks[buf][(j * 16) * 72], 72, lane);
      v16h kb1 = load_frag(&Ks[buf][(j * 16) * 72 + 32], 72, lane);
      sc[j] = wmma_f16(q0, kb0, sc[j]);
      sc[j] = wmma_f16(q1, kb1, sc[j]);
    }

    // causal mask + online softmax
    float nm[8];
#pragma unroll
    for (int j = 0; j < 4; ++j)
#pragma unroll
      for (int r = 0; r < 8; ++r) {
        int col = s0 + j * 16 + cbase;
        float v = sc[j][r];
        if (col > rowg[r]) v = -1e30f;
        sc[j][r] = v;
      }
#pragma unroll
    for (int r = 0; r < 8; ++r) {
      float mx = fmaxf(fmaxf(sc[0][r], sc[1][r]), fmaxf(sc[2][r], sc[3][r]));
      nm[r] = fmaxf(mr[r], redmax16(mx));
    }
    float alpha[8];
#pragma unroll
    for (int r = 0; r < 8; ++r) {
      alpha[r] = __expf(mr[r] - nm[r]);
      float s = 0.f;
#pragma unroll
      for (int j = 0; j < 4; ++j) {
        float p = __expf(sc[j][r] - nm[r]);
        sc[j][r] = p;
        s += p;
      }
      sr[r] = sr[r] * alpha[r] + redsum16(s);
      mr[r] = nm[r];
    }
#pragma unroll
    for (int j = 0; j < 4; ++j)
#pragma unroll
      for (int r = 0; r < 8; ++r) o[j][r] *= alpha[r];

    // transpose P (C-layout) -> A-layout via wave-private LDS region
#pragma unroll
    for (int j = 0; j < 4; ++j)
#pragma unroll
      for (int r = 0; r < 8; ++r)
        Ps[(w * 16 + r + rbase) * 72 + j * 16 + cbase] = (_Float16)sc[j][r];

    v16h pa0 = load_frag(&Ps[(w * 16) * 72], 72, lane);
    v16h pa1 = load_frag(&Ps[(w * 16) * 72 + 32], 72, lane);

    // O += P * V  (B-fragment lane = e, rows of Vs are e -> contiguous over s)
#pragma unroll
    for (int j = 0; j < 4; ++j) {
      v16h vb0 = load_frag(&Vs[buf][(j * 16) * 72], 72, lane);
      v16h vb1 = load_frag(&Vs[buf][(j * 16) * 72 + 32], 72, lane);
      o[j] = wmma_f16(pa0, vb0, o[j]);
      o[j] = wmma_f16(pa1, vb1, o[j]);
    }
  }

  // normalize and write O in [b][l][h*E+e] f16 for the output projection
#pragma unroll
  for (int j = 0; j < 4; ++j)
#pragma unroll
    for (int r = 0; r < 8; ++r) {
      int e = j * 16 + cbase;
      float v = o[j][r] / sr[r];
      Oc[((size_t)b * LL + rowg[r]) * DD + h * EE + e] = (_Float16)v;
    }
}

extern "C" void kernel_launch(void* const* d_in, const int* in_sizes, int n_in,
                              void* d_out, int out_size, void* d_ws, size_t ws_size,
                              hipStream_t stream) {
  const float* q  = (const float*)d_in[0];
  const float* k  = (const float*)d_in[1];
  const float* v  = (const float*)d_in[2];
  const float* Wq = (const float*)d_in[3];
  const float* Wk = (const float*)d_in[4];
  const float* Wv = (const float*)d_in[5];
  const float* Wo = (const float*)d_in[6];
  float* out = (float*)d_out;

  _Float16* ws  = (_Float16*)d_ws;
  _Float16* Wqt = ws;                    // 512*512
  _Float16* Wkt = Wqt + 262144;
  _Float16* Wvt = Wkt + 262144;
  _Float16* Wot = Wvt + 262144;
  _Float16* Qh  = Wot + 262144;          // B*L*D
  _Float16* Kh  = Qh + 4194304;
  _Float16* Vt  = Kh + 4194304;
  _Float16* Oc  = Vt + 4194304;

  const float SCALE = 0.125f;  // 1/sqrt(64)

  k_wT<<<dim3(1024), dim3(256), 0, stream>>>(Wq, Wqt);
  k_wT<<<dim3(1024), dim3(256), 0, stream>>>(Wk, Wkt);
  k_wT<<<dim3(1024), dim3(256), 0, stream>>>(Wv, Wvt);
  k_wT<<<dim3(1024), dim3(256), 0, stream>>>(Wo, Wot);

  dim3 gg(DD / 128, (BB * LL) / 128);    // (4, 64)
  k_proj<<<gg, dim3(256), 0, stream>>>(q, Wqt, Qh, SCALE, 0);
  k_proj<<<gg, dim3(256), 0, stream>>>(k, Wkt, Kh, 1.0f, 0);
  k_proj<<<gg, dim3(256), 0, stream>>>(v, Wvt, Vt, 1.0f, 1);

  k_attn<<<dim3(LL / 64, BB * HH), dim3(128), 0, stream>>>(Qh, Kh, Vt, Oc);

  k_oproj<<<gg, dim3(256), 0, stream>>>(Oc, Wot, out);
}
